// LSTM_42485816492102
// MI455X (gfx1250) — matrix operand order repaired and checked by reading, not compiled, not run on hardware
//
#include <hip/hip_runtime.h>
#include <hip/hip_bf16.h>

// ---------------- problem constants ----------------
#define Bq 64
#define Sq 512
#define Hq 1024
#define Lq 2
#define JW 16                 // h-columns owned per workgroup
#define NWG (Hq / JW)         // 64 workgroups per layer kernel
#define KT (Hq / 32)          // 32 k-tiles of K=32
#define FRAG 512              // bf16 elements per 16x32 / 32x16 fragment
#define THREADS 256           // 8 wave32 waves
#define WAVES 8

typedef __bf16 bf16_t;
typedef __attribute__((ext_vector_type(4)))  __bf16 v4bf;
typedef __attribute__((ext_vector_type(8)))  __bf16 v8bf;
typedef __attribute__((ext_vector_type(16))) __bf16 v16bf;
typedef __attribute__((ext_vector_type(4)))  float  v4f;
typedef __attribute__((ext_vector_type(8)))  float  v8f;

__device__ __forceinline__ float sigmoidf_(float x) {
    return 1.0f / (1.0f + __expf(-x));
}

// ---------------------------------------------------------------------------
// Pack Wi/Wh (f32, [L,4,H,H] = [gate][k_in][n_out]) into bf16 WMMA B-fragments.
// Fragment layout (32x16 B tile, 16-bit): lane<16 holds col n=lane, K=k0..k0+15;
// lane>=16 holds col n=lane-16, K=k0+16..k0+31; 16 bf16 contiguous per lane.
// Global order: [l][wt][g][j][kt][lane][i]  (wt: 0=Wi, 1=Wh)
// ---------------------------------------------------------------------------
__global__ void pack_weights_kernel(const float* __restrict__ Wi,
                                    const float* __restrict__ Wh,
                                    bf16_t* __restrict__ out) {
    const long long total = (long long)Lq * 2 * 4 * NWG * KT * FRAG;
    for (long long idx = (long long)blockIdx.x * blockDim.x + threadIdx.x;
         idx < total; idx += (long long)gridDim.x * blockDim.x) {
        int e = (int)(idx % FRAG);
        long long r = idx / FRAG;
        int kt = (int)(r % KT); r /= KT;
        int j  = (int)(r % NWG); r /= NWG;
        int g  = (int)(r % 4);  r /= 4;
        int wt = (int)(r % 2);  r /= 2;
        int l  = (int)r;
        int lane = e >> 4;
        int i    = e & 15;
        int k = kt * 32 + ((lane >= 16) ? 16 : 0) + i;
        int n = j * JW + (lane & 15);
        const float* W = wt ? Wh : Wi;
        out[idx] = (bf16_t)W[((long long)(l * 4 + g) * Hq + k) * Hq + n];
    }
}

// fp32 -> bf16 elementwise (x sequence)
__global__ void pack_x_kernel(const float* __restrict__ x,
                              bf16_t* __restrict__ out, long long n) {
    for (long long i = (long long)blockIdx.x * blockDim.x + threadIdx.x;
         i < n; i += (long long)gridDim.x * blockDim.x) {
        out[i] = (bf16_t)x[i];
    }
}

// ---------------------------------------------------------------------------
// One K-reduction for a timestep: branch-free (HAS_H templated), with the
// A-fragment global loads software-pipelined one k-tile ahead so the WMMAs
// for k-tile kt overlap the loads for kt+1.
// ---------------------------------------------------------------------------
template <bool HAS_H>
__device__ __forceinline__ void lstm_matmul(const bf16_t* __restrict__ xrow,
                                            const bf16_t* __restrict__ hrow,
                                            const bf16_t* __restrict__ w_lds,
                                            int gb, int lane, int ksub,
                                            v8f& acc0, v8f& acc1) {
    const bf16_t* bbase = w_lds + lane * 16;
    v8bf xlo = *(const v8bf*)(xrow + ksub);
    v8bf xhi = *(const v8bf*)(xrow + ksub + 8);
    v8bf hlo = {}, hhi = {};
    if (HAS_H) {
        hlo = *(const v8bf*)(hrow + ksub);
        hhi = *(const v8bf*)(hrow + ksub + 8);
    }
    for (int kt = 0; kt < KT; ++kt) {
        // issue next k-tile's global A loads before consuming this one
        v8bf xlo_n = xlo, xhi_n = xhi, hlo_n = hlo, hhi_n = hhi;
        if (kt + 1 < KT) {
            const int kcn = (kt + 1) * 32 + ksub;
            xlo_n = *(const v8bf*)(xrow + kcn);
            xhi_n = *(const v8bf*)(xrow + kcn + 8);
            if (HAS_H) {
                hlo_n = *(const v8bf*)(hrow + kcn);
                hhi_n = *(const v8bf*)(hrow + kcn + 8);
            }
        }
        const v16bf ax = __builtin_shufflevector(xlo, xhi,
            0,1,2,3,4,5,6,7,8,9,10,11,12,13,14,15);
        const bf16_t* bk = bbase + (long long)kt * FRAG;
        const v16bf b0 = *(const v16bf*)(bk + (long long)(gb)     * KT * FRAG);
        const v16bf b1 = *(const v16bf*)(bk + (long long)(gb + 1) * KT * FRAG);
        acc0 = __builtin_amdgcn_wmma_f32_16x16x32_bf16(false, ax, false, b0, (short)0, acc0, false, false);
        acc1 = __builtin_amdgcn_wmma_f32_16x16x32_bf16(false, ax, false, b1, (short)0, acc1, false, false);
        if (HAS_H) {
            const v16bf ah = __builtin_shufflevector(hlo, hhi,
                0,1,2,3,4,5,6,7,8,9,10,11,12,13,14,15);
            const v16bf b2 = *(const v16bf*)(bk + (long long)(4 + gb)     * KT * FRAG);
            const v16bf b3 = *(const v16bf*)(bk + (long long)(4 + gb + 1) * KT * FRAG);
            acc0 = __builtin_amdgcn_wmma_f32_16x16x32_bf16(false, ah, false, b2, (short)0, acc0, false, false);
            acc1 = __builtin_amdgcn_wmma_f32_16x16x32_bf16(false, ah, false, b3, (short)0, acc1, false, false);
        }
        xlo = xlo_n; xhi = xhi_n;
        if (HAS_H) { hlo = hlo_n; hhi = hhi_n; }
    }
}

// ---------------------------------------------------------------------------
// Persistent per-layer LSTM kernel. gridDim.x == NWG, blockDim.x == 256.
// Workgroup j owns h-columns [j*16, j*16+16). Weight slices live in LDS for
// the entire sequence. Grid-wide barrier per timestep via atomic counter.
// ---------------------------------------------------------------------------
struct StepCtx {
    const bf16_t* seq_in;
    bf16_t* seq_out;
    bf16_t* hbuf;
    unsigned* syncc;
    float* dout;
    const bf16_t* w_lds;
    float* stage;
    float* cstate;
    const float* biasld;
    int j, layer, is_last;
};

template <bool HAS_H>
__device__ __forceinline__ void lstm_step(const StepCtx& cx, int t) {
    const int tid  = threadIdx.x;
    const int lane = tid & 31;
    const int w    = tid >> 5;
    const int mtile = w & 3;
    const int gb    = (w >> 2) * 2;
    const int arow  = mtile * 16 + (lane & 15);
    const int ksub  = (lane >= 16) ? 16 : 0;
    const int ncol  = lane & 15;

    v8f acc0 = {}; v8f acc1 = {};
    const bf16_t* xrow = cx.seq_in + ((long long)arow * Sq + t) * Hq;
    const bf16_t* hrow = cx.hbuf + (long long)(t & 1) * Bq * Hq + (long long)arow * Hq;
    if (t + 1 < Sq)  // warm caches for next timestep's activations
        __builtin_prefetch(cx.seq_in + ((long long)arow * Sq + (t + 1)) * Hq, 0, 1);

    lstm_matmul<HAS_H>(xrow, hrow, cx.w_lds, gb, lane, ksub, acc0, acc1);

    // bias + pre-activations to LDS staging. C/D layout:
    // lane l, vgpr r -> (M = r + 8*(l>=16), N = l&15)
    const float bs0 = cx.biasld[gb * 16 + ncol];
    const float bs1 = cx.biasld[(gb + 1) * 16 + ncol];
    const int mbase = mtile * 16 + (lane >> 4) * 8;
    #pragma unroll
    for (int r = 0; r < 8; ++r) {
        cx.stage[(gb)     * (Bq * JW) + (mbase + r) * JW + ncol] = acc0[r] + bs0;
        cx.stage[(gb + 1) * (Bq * JW) + (mbase + r) * JW + ncol] = acc1[r] + bs1;
    }
    __syncthreads();

    // elementwise gate math: 1024 (b, col) cells, 4 contiguous per thread
    const int p0  = tid * 4;
    const int b   = p0 >> 4;
    const int nn0 = p0 & 15;
    float hv[4], cv[4];
    #pragma unroll
    for (int e = 0; e < 4; ++e) {
        const int p = p0 + e;
        const float gi = sigmoidf_(cx.stage[0 * (Bq * JW) + p]);
        const float gf = sigmoidf_(cx.stage[1 * (Bq * JW) + p]);
        const float gg = tanhf    (cx.stage[2 * (Bq * JW) + p]);
        const float go = sigmoidf_(cx.stage[3 * (Bq * JW) + p]);
        const float c  = gf * cx.cstate[p] + gi * gg;
        const float h  = go * tanhf(c);
        cx.cstate[p] = c;
        hv[e] = h; cv[e] = c;
    }
    const int col0 = cx.j * JW + nn0;
    const v4bf hb = { (bf16_t)hv[0], (bf16_t)hv[1], (bf16_t)hv[2], (bf16_t)hv[3] };
    *(v4bf*)(cx.hbuf + (long long)((t + 1) & 1) * Bq * Hq + (long long)b * Hq + col0) = hb;
    *(v4bf*)(cx.seq_out + ((long long)b * Sq + t) * Hq + col0) = hb;
    if (t == Sq - 1) {
        const v4f hf = { hv[0], hv[1], hv[2], hv[3] };
        const v4f cf = { cv[0], cv[1], cv[2], cv[3] };
        *(v4f*)(cx.dout + (long long)(1 + cx.layer) * Bq * Hq + (long long)b * Hq + col0) = hf; // h_n
        *(v4f*)(cx.dout + (long long)(3 + cx.layer) * Bq * Hq + (long long)b * Hq + col0) = cf; // c_n
        if (cx.is_last)
            *(v4f*)(cx.dout + (long long)b * Hq + col0) = hf;                                   // out0
    }
    __threadfence();
    __syncthreads();
    if (tid == 0) {  // grid barrier: all WGs finished step t
        atomicAdd(cx.syncc, 1u);
        const unsigned target = (unsigned)(NWG * (t + 1));
        while (__hip_atomic_load(cx.syncc, __ATOMIC_ACQUIRE, __HIP_MEMORY_SCOPE_AGENT) < target)
            __builtin_amdgcn_s_sleep(2);
    }
    __syncthreads();
}

__global__ void lstm_layer_kernel(const bf16_t* __restrict__ seq_in,   // [B,S,H] bf16
                                  bf16_t* __restrict__ seq_out,        // [B,S,H] bf16
                                  const bf16_t* __restrict__ wpack,    // this layer's packed W
                                  const float* __restrict__ bi_l,      // [4,H]
                                  const float* __restrict__ bh_l,      // [4,H]
                                  bf16_t* __restrict__ hbuf,           // [2][B*H] ping-pong
                                  unsigned* __restrict__ syncc,        // grid barrier counter
                                  float* __restrict__ dout,            // full output base
                                  int layer, int is_last) {
    extern __shared__ char smem_raw[];
    bf16_t* w_lds  = (bf16_t*)smem_raw;                                   // 256KB
    float*  stage  = (float*)(smem_raw + (size_t)2 * 4 * KT * FRAG * 2);  // 4*1024 f32
    float*  cstate = stage + 4 * Bq * JW;                                 // 1024 f32
    float*  biasld = cstate + Bq * JW;                                    // 64 f32

    const int tid = threadIdx.x;
    const int j   = blockIdx.x;

    // ---- stage weight slices into LDS (8 contiguous 32KB runs) ----
    for (int wg = 0; wg < 8; ++wg) {
        const uint4* src = (const uint4*)(wpack + ((long long)(wg * NWG + j) * KT) * FRAG);
        uint4* dst = (uint4*)(w_lds + (long long)wg * KT * FRAG);
        for (int it = tid; it < KT * FRAG / 8; it += THREADS) dst[it] = src[it];
    }
    if (tid < 64) {  // fused bias (bi + bh) for our 4x16 gate columns
        int g = tid >> 4, n = tid & 15;
        biasld[tid] = bi_l[g * Hq + j * JW + n] + bh_l[g * Hq + j * JW + n];
    }
    for (int p = tid; p < Bq * JW; p += THREADS) cstate[p] = 0.0f;
    __syncthreads();

    StepCtx cx = { seq_in, seq_out, hbuf, syncc, dout,
                   w_lds, stage, cstate, biasld, j, layer, is_last };

    lstm_step<false>(cx, 0);                      // t=0: h == 0, x-side only
    for (int t = 1; t < Sq; ++t)                  // steady state: branch-free inner loop
        lstm_step<true>(cx, t);
}

// ---------------------------------------------------------------------------
extern "C" void kernel_launch(void* const* d_in, const int* in_sizes, int n_in,
                              void* d_out, int out_size, void* d_ws, size_t ws_size,
                              hipStream_t stream) {
    const float* x  = (const float*)d_in[0];
    const float* Wi = (const float*)d_in[1];
    const float* bi = (const float*)d_in[2];
    const float* Wh = (const float*)d_in[3];
    const float* bh = (const float*)d_in[4];
    float* out = (float*)d_out;

    // workspace carve (bf16 halves)
    bf16_t* wpack = (bf16_t*)d_ws;
    const long long wpackN = (long long)Lq * 2 * 4 * NWG * KT * FRAG;  // 16,777,216 (32MB)
    const long long seqN   = (long long)Bq * Sq * Hq;                  // 33,554,432 (64MB ea)
    bf16_t* xbf  = wpack + wpackN;
    bf16_t* seq1 = xbf + seqN;
    bf16_t* hbuf = seq1 + seqN;                                        // 2*B*H bf16
    unsigned* syncc = (unsigned*)(hbuf + 2LL * Bq * Hq);

    const size_t smem = (size_t)2 * 4 * KT * FRAG * 2                 // weights (bf16)
                      + (size_t)(4 * Bq * JW + Bq * JW + 64) * 4;     // stage + cstate + bias
    (void)hipFuncSetAttribute((const void*)lstm_layer_kernel,
                              hipFuncAttributeMaxDynamicSharedMemorySize, (int)smem);

    pack_weights_kernel<<<4096, THREADS, 0, stream>>>(Wi, Wh, wpack);
    pack_x_kernel<<<4096, THREADS, 0, stream>>>(x, xbf, seqN);

    // layer 0: x -> seq1
    hipMemsetAsync(syncc, 0, sizeof(unsigned), stream);
    lstm_layer_kernel<<<NWG, THREADS, smem, stream>>>(
        xbf, seq1, wpack, bi, bh, hbuf, syncc, out, /*layer=*/0, /*is_last=*/0);

    // layer 1: seq1 -> (scratch, reuse xbf region)
    hipMemsetAsync(syncc, 0, sizeof(unsigned), stream);
    lstm_layer_kernel<<<NWG, THREADS, smem, stream>>>(
        seq1, xbf, wpack + (long long)2 * 4 * NWG * KT * FRAG,
        bi + 4 * Hq, bh + 4 * Hq, hbuf, syncc, out, /*layer=*/1, /*is_last=*/1);
}
